// LocationAwareAttention_9208409883132
// MI455X (gfx1250) — matrix operand order repaired and checked by reading, not compiled, not run on hardware
//
#include <hip/hip_runtime.h>
#include <hip/hip_bf16.h>
#include <math.h>

typedef __attribute__((ext_vector_type(16))) __bf16 v16bf;
typedef __attribute__((ext_vector_type(8)))  __bf16 v8bf;
typedef __attribute__((ext_vector_type(8)))  float  v8f;

#define B_SZ    16
#define KLEN    4096
#define DM      1024
#define HEADS   8
#define DIM     128
#define KERNELS 10

// ---------------------------------------------------------------------------
// Kernel 0: w_vT[e][f] = (bf16) w_v[f][e]   (transpose + convert, 2 MB out,
// stays L2-resident for the whole GEMM)
// ---------------------------------------------------------------------------
__global__ __launch_bounds__(256)
void k_wvT(const float* __restrict__ w_v, __bf16* __restrict__ wvT) {
  int idx = blockIdx.x * 256 + threadIdx.x;     // 1M elements total
  int e = idx >> 10;
  int f = idx & 1023;
  wvT[(size_t)e * DM + f] = (__bf16)w_v[(size_t)f * DM + e];
}

// ---------------------------------------------------------------------------
// Kernel 1: qp[b][e] = sum_f query[b][f] * w_q[f][e]     (16x1024 outputs)
// ---------------------------------------------------------------------------
__global__ __launch_bounds__(256)
void k_qproj(const float* __restrict__ query, const float* __restrict__ w_q,
             float* __restrict__ qp) {
  const int b  = blockIdx.x;
  const int e0 = threadIdx.x;                   // 4 outputs per thread
  const float* q = query + (size_t)b * DM;
  float a0 = 0.f, a1 = 0.f, a2 = 0.f, a3 = 0.f;
  for (int f = 0; f < DM; ++f) {
    float qf = q[f];
    const float* w = w_q + (size_t)f * DM;
    a0 += qf * w[e0];
    a1 += qf * w[e0 + 256];
    a2 += qf * w[e0 + 512];
    a3 += qf * w[e0 + 768];
  }
  float* o = qp + (size_t)b * DM;
  o[e0] = a0; o[e0 + 256] = a1; o[e0 + 512] = a2; o[e0 + 768] = a3;
}

// ---------------------------------------------------------------------------
// Kernel 2 (main): per (b, 16-key tile) workgroup, 8 waves = 8 heads.
// Wave w computes kproj[16 x 128] of head w via WMMA bf16 and fuses the
// additive-attention epilogue (tanh + score dot + cross-lane reduce).
// ---------------------------------------------------------------------------
__global__ __launch_bounds__(256)
void k_main(const float* __restrict__ key_in, const float* __restrict__ prev_align,
            const float* __restrict__ conv_w, const float* __restrict__ conv_b,
            const float* __restrict__ w_u,    const float* __restrict__ bias,
            const float* __restrict__ score_w,const float* __restrict__ score_b,
            const float* __restrict__ qp,     const __bf16* __restrict__ wvT,
            float* __restrict__ scores) {
  __shared__ __bf16 sA[16][DM];          // key tile, bf16 (32 KB)
  __shared__ float  sLocE[16][DIM];      // loc energy (8 KB), head-independent
  __shared__ float  sLC[16][KERNELS];    // conv output per key

  const int b     = blockIdx.y;
  const int kbase = blockIdx.x * 16;
  const int tid   = threadIdx.x;
  const int lane  = tid & 31;
  const int wave  = tid >> 5;            // == head index
  const int half  = lane >> 4;           // 0: lanes 0-15, 1: lanes 16-31
  const int l16   = lane & 15;

  // --- conv1d (pad=1, cross-correlation) over heads -> sLC[k][c] ---
  if (tid < 16 * KERNELS) {
    int k = tid / KERNELS, c = tid % KERNELS;
    int kg = kbase + k;
    float s = conv_b[c];
#pragma unroll
    for (int h = 0; h < HEADS; ++h) {
      const float* pa = prev_align + ((size_t)b * HEADS + h) * KLEN;
      float x0 = (kg > 0)        ? pa[kg - 1] : 0.f;
      float x1 = pa[kg];
      float x2 = (kg < KLEN - 1) ? pa[kg + 1] : 0.f;
      const float* w = conv_w + ((size_t)c * HEADS + h) * 3;
      s += x0 * w[0] + x1 * w[1] + x2 * w[2];
    }
    sLC[k][c] = s;
  }

  // --- stage key tile into LDS as bf16 (coalesced float4 reads) ---
  {
    const float* kb = key_in + ((size_t)b * KLEN + kbase) * DM;
    for (int i = tid; i < 16 * (DM / 4); i += 256) {
      int r  = i >> 8;                   // 256 float4 per row
      int c4 = (i & 255) << 2;
      float4 v = *(const float4*)(kb + (size_t)r * DM + c4);
      sA[r][c4 + 0] = (__bf16)v.x;
      sA[r][c4 + 1] = (__bf16)v.y;
      sA[r][c4 + 2] = (__bf16)v.z;
      sA[r][c4 + 3] = (__bf16)v.w;
    }
  }
  __syncthreads();

  // --- loc_energy[k][d] = tanh(sum_c sLC[k][c] * w_u[c][d]) ---
  for (int i = tid; i < 16 * DIM; i += 256) {
    int k = i >> 7, d = i & 127;
    float s = 0.f;
#pragma unroll
    for (int c = 0; c < KERNELS; ++c) s += sLC[k][c] * w_u[(size_t)c * DIM + d];
    sLocE[k][d] = tanhf(s);
  }
  __syncthreads();

  // --- WMMA GEMM: kproj tile = sA(16x1024) x wvT(head slice, 1024x128) ---
  v8f acc[8] = {};
  const __bf16* bcolbase = wvT + ((size_t)wave * DIM + l16) * DM + half * 8;

  for (int kk = 0; kk < DM; kk += 32) {
    // A fragment (ISA 16-bit A 16x32 layout):
    //  lanes 0-15: row=l16, K = kk+{0..7,16..23}; lanes 16-31: K = kk+{8..15,24..31}
    const __bf16* ap = &sA[l16][kk + half * 8];
    v8bf a0 = *(const v8bf*)ap;
    v8bf a1 = *(const v8bf*)(ap + 16);
    v16bf afrag;
#pragma unroll
    for (int i = 0; i < 8; ++i) { afrag[i] = a0[i]; afrag[i + 8] = a1[i]; }

#pragma unroll
    for (int t = 0; t < 8; ++t) {
      // B fragment: col N = 16t + l16; lanes 0-15: K = kk+{0..7,16..23},
      // lanes 16-31: K = kk+{8..15,24..31} (matches A layout)
      const __bf16* bp = bcolbase + (size_t)t * 16 * DM + kk;
      v8bf b0 = *(const v8bf*)bp;
      v8bf b1 = *(const v8bf*)(bp + 16);
      v16bf bfrag;
#pragma unroll
      for (int i = 0; i < 8; ++i) { bfrag[i] = b0[i]; bfrag[i + 8] = b1[i]; }

      acc[t] = __builtin_amdgcn_wmma_f32_16x16x32_bf16(
          false, afrag, false, bfrag, (short)0, acc[t], false, false);
    }
  }

  // --- fused epilogue: e = tanh(kproj + q + locE + bias); score = e . score_w
  // C/D layout: VGPR j, lane: row m = j + 8*half, col n = l16 (tile t -> d=16t+n)
  const float sb = score_b[0];
  float p[8] = {0.f, 0.f, 0.f, 0.f, 0.f, 0.f, 0.f, 0.f};
#pragma unroll
  for (int t = 0; t < 8; ++t) {
    int d = 16 * t + l16;
    float qv = qp[((size_t)b * HEADS + wave) * DIM + d];
    float sw = score_w[d];
    float bi = bias[d];
#pragma unroll
    for (int j = 0; j < 8; ++j) {
      int m = j + 8 * half;
      float e = tanhf(acc[t][j] + qv + sLocE[m][d] + bi);
      p[j] += e * sw;
    }
  }
#pragma unroll
  for (int j = 0; j < 8; ++j) {
    float v = p[j];
    v += __shfl_xor(v, 1, 32);
    v += __shfl_xor(v, 2, 32);
    v += __shfl_xor(v, 4, 32);
    v += __shfl_xor(v, 8, 32);          // reduce over the 16 lanes of this half
    if (l16 == 0) {
      int m = j + 8 * half;
      scores[((size_t)b * HEADS + wave) * KLEN + kbase + m] = v + sb;
    }
  }
}

// ---------------------------------------------------------------------------
// Kernel 3: softmax over K=4096 per (b,h) row
// ---------------------------------------------------------------------------
__global__ __launch_bounds__(256)
void k_softmax(const float* __restrict__ scores, float* __restrict__ align) {
  __shared__ float red[8];
  const int bh  = blockIdx.x;
  const int tid = threadIdx.x;
  const float* row = scores + (size_t)bh * KLEN;
  float* out = align + (size_t)bh * KLEN;

  float m = -1e30f;
  for (int i = tid; i < KLEN; i += 256) m = fmaxf(m, row[i]);
  for (int o = 1; o < 32; o <<= 1) m = fmaxf(m, __shfl_xor(m, o, 32));
  if ((tid & 31) == 0) red[tid >> 5] = m;
  __syncthreads();
  m = red[0];
#pragma unroll
  for (int i = 1; i < 8; ++i) m = fmaxf(m, red[i]);
  __syncthreads();

  float s = 0.f;
  for (int i = tid; i < KLEN; i += 256) {
    float e = __expf(row[i] - m);
    out[i] = e;
    s += e;
  }
  for (int o = 1; o < 32; o <<= 1) s += __shfl_xor(s, o, 32);
  if ((tid & 31) == 0) red[tid >> 5] = s;
  __syncthreads();
  s = red[0];
#pragma unroll
  for (int i = 1; i < 8; ++i) s += red[i];
  float inv = 1.f / s;
  for (int i = tid; i < KLEN; i += 256) out[i] *= inv;
}

// ---------------------------------------------------------------------------
// Kernel 4: context[b*H+h][d] = sum_k align[b,h,k] * value[b,k,h*128+d]
// (streams `value` once; HBM-bound)
// ---------------------------------------------------------------------------
__global__ __launch_bounds__(128)
void k_context(const float* __restrict__ align, const float* __restrict__ value,
               float* __restrict__ context) {
  const int bh = blockIdx.x;
  const int b  = bh >> 3, h = bh & 7;
  const int d  = threadIdx.x;
  const float* al = align + (size_t)bh * KLEN;
  const float* vp = value + (size_t)b * KLEN * DM + h * DIM + d;
  float acc = 0.f;
  for (int k = 0; k < KLEN; k += 4) {
    acc += al[k + 0] * vp[(size_t)(k + 0) * DM];
    acc += al[k + 1] * vp[(size_t)(k + 1) * DM];
    acc += al[k + 2] * vp[(size_t)(k + 2) * DM];
    acc += al[k + 3] * vp[(size_t)(k + 3) * DM];
  }
  context[(size_t)bh * DIM + d] = acc;
}

// ---------------------------------------------------------------------------
extern "C" void kernel_launch(void* const* d_in, const int* in_sizes, int n_in,
                              void* d_out, int out_size, void* d_ws, size_t ws_size,
                              hipStream_t stream) {
  (void)in_sizes; (void)n_in; (void)out_size; (void)ws_size;
  const float* query      = (const float*)d_in[0];
  const float* key_in     = (const float*)d_in[1];
  const float* value      = (const float*)d_in[2];
  const float* prev_align = (const float*)d_in[3];
  const float* conv_w     = (const float*)d_in[4];
  const float* conv_b     = (const float*)d_in[5];
  const float* w_u        = (const float*)d_in[6];
  const float* w_q        = (const float*)d_in[7];
  const float* w_v        = (const float*)d_in[8];
  const float* bias       = (const float*)d_in[9];
  const float* score_w    = (const float*)d_in[10];
  const float* score_b    = (const float*)d_in[11];

  float* out     = (float*)d_out;
  float* context = out;                          // [B*H, DIM] = 16384 floats
  float* align   = out + B_SZ * HEADS * DIM;     // [B, H, K]

  char* ws = (char*)d_ws;
  float*  qp     = (float*)ws;                                        // 64 KB
  __bf16* wvT    = (__bf16*)(ws + 64 * 1024);                         // 2 MB
  float*  scores = (float*)(ws + 64 * 1024 + 2 * 1024 * 1024);        // 2 MB

  k_wvT    <<<dim3(DM * DM / 256),      256, 0, stream>>>(w_v, wvT);
  k_qproj  <<<dim3(B_SZ),               256, 0, stream>>>(query, w_q, qp);
  k_main   <<<dim3(KLEN / 16, B_SZ),    256, 0, stream>>>(key_in, prev_align,
                conv_w, conv_b, w_u, bias, score_w, score_b, qp, wvT, scores);
  k_softmax<<<dim3(B_SZ * HEADS),       256, 0, stream>>>(scores, align);
  k_context<<<dim3(B_SZ * HEADS),       128, 0, stream>>>(align, value, context);
}
